// NaiveTransformerLayer_69441031241819
// MI455X (gfx1250) — compile-verified, matrix operand order checked by reading, not run on hardware
//
#include <hip/hip_runtime.h>

// ---------------------------------------------------------------------------
// Types for WMMA bf16 (gfx1250): A/B = 16 x bf16 (8 VGPRs), C/D = 8 x f32.
// ---------------------------------------------------------------------------
typedef __attribute__((ext_vector_type(16))) __bf16 v16bf;
typedef __attribute__((ext_vector_type(8)))  float  v8f;

union FragAB {
    v16bf v;
    unsigned int u[8];
};

__device__ __forceinline__ unsigned short f2bf(float f) {
    unsigned int u = __float_as_uint(f);
    unsigned int r = u + 0x7FFFu + ((u >> 16) & 1u);   // round-to-nearest-even
    return (unsigned short)(r >> 16);
}

// ---------------------------------------------------------------------------
// Elementwise f32 -> bf16 cast
// ---------------------------------------------------------------------------
__global__ __launch_bounds__(256) void cast_f32_bf16(const float* __restrict__ in,
                                                     unsigned short* __restrict__ out,
                                                     long long n) {
    long long i = (long long)blockIdx.x * blockDim.x + threadIdx.x;
    long long stride = (long long)gridDim.x * blockDim.x;
    for (; i < n; i += stride) out[i] = f2bf(in[i]);
}

// ---------------------------------------------------------------------------
// bf16 WMMA GEMM: C[M,N] = act(A[M,K] @ op(B) + bias)
//   op(B) = B [K,N] row-major       (transB==0, ldb==N)
//   op(B) = B'[N,K] row-major (^T)  (transB==1, ldb==K)   -> Q @ K^T
//
// Workgroup tile 128x128, BK=32, 8 waves; wave tile 32x64 = 2x4 WMMA frags
// (8 v_wmma per wave per K-step). A and B tiles live in LDS in the
// *fragment-native* [row][k] layout (B stored transposed) so every fragment
// is exactly two aligned ds_load_b128.
//
// DOUBLE-BUFFERED: stage tile kk+32 into buf^1 while computing from buf;
// one barrier per K-step. global_prefetch for the kk+64 tile.
// Batched via gridDim.z with element strides sA/sB/sC.
// ---------------------------------------------------------------------------
#define LDK 40   // padded k-stride in halves; rows are 80B (16B aligned)

__device__ __forceinline__ void load_frag(const unsigned short* __restrict__ lds,
                                          int row, int lhalf, FragAB& f) {
    const uint4 lo = *(const uint4*)&lds[row * LDK + lhalf * 8];
    const uint4 hi = *(const uint4*)&lds[row * LDK + 16 + lhalf * 8];
    f.u[0] = lo.x; f.u[1] = lo.y; f.u[2] = lo.z; f.u[3] = lo.w;
    f.u[4] = hi.x; f.u[5] = hi.y; f.u[6] = hi.z; f.u[7] = hi.w;
}

__global__ __launch_bounds__(256) void gemm_bf16_wmma(
    const unsigned short* __restrict__ A, const unsigned short* __restrict__ B,
    const float* __restrict__ bias, float* __restrict__ Cf,
    unsigned short* __restrict__ Cb,
    int M, int N, int K, int ldb, int transB, int actGelu,
    long long sA, long long sB, long long sC)
{
    __shared__ unsigned short Alds[2][128 * LDK];   // [m][k]
    __shared__ unsigned short Blds[2][128 * LDK];   // [n][k]  (B transposed)

    const int bz = blockIdx.z;
    A += (long long)bz * sA;
    B += (long long)bz * sB;
    const long long cOff = (long long)bz * sC;

    const int m0 = blockIdx.y * 128;
    const int n0 = blockIdx.x * 128;

    const int t     = threadIdx.x;
    const int lane  = t & 31;
    const int w     = t >> 5;
    const int lhalf = lane >> 4;     // 0/1 : lanes 0-15 vs 16-31
    const int lmod  = lane & 15;
    const int wrow  = (w & 3) * 32;  // 4 wave-rows of 32
    const int wcol  = (w >> 2) * 64; // 2 wave-cols of 64

    // Per-thread staging coordinates (invariant across K-steps).
    const int arow  = t >> 1;
    const int ahalf = (t & 1) * 16;
    // transB==1 staging
    const int tn    = t >> 1;
    const int tseg  = (t & 1) * 16;
    // transB==0 staging (k-pair packing transpose)
    const int kp    = (t >> 4) * 2;
    const int nbase = (t & 15) * 8;

    // Stage tile for K-offset kk into buffer buf.
    auto stage = [&](int kk, int buf) {
        {   // A tile: 128 rows x 32 halves, 2 threads/row, 32B each
            const unsigned short* src = A + (long long)(m0 + arow) * K + kk + ahalf;
            const uint4 q0 = ((const uint4*)src)[0];
            const uint4 q1 = ((const uint4*)src)[1];
            uint4* dst = (uint4*)&Alds[buf][arow * LDK + ahalf];
            dst[0] = q0; dst[1] = q1;
            if (kk + 64 < K) __builtin_prefetch(src + 64, 0, 0);
        }
        if (transB) {
            // B' is [N,K] row-major == [n][k] already: straight b128 copy
            const unsigned short* src = B + (long long)(n0 + tn) * ldb + kk + tseg;
            const uint4 q0 = ((const uint4*)src)[0];
            const uint4 q1 = ((const uint4*)src)[1];
            uint4* dst = (uint4*)&Blds[buf][tn * LDK + tseg];
            dst[0] = q0; dst[1] = q1;
            if (kk + 64 < K) __builtin_prefetch(src + 64, 0, 0);
        } else {
            // B is [K,N] row-major: transpose via register k-pair packing.
            const unsigned short* r0p = B + (long long)(kk + kp    ) * ldb + n0 + nbase;
            const unsigned short* r1p = B + (long long)(kk + kp + 1) * ldb + n0 + nbase;
            const uint4 q0 = *(const uint4*)r0p;
            const uint4 q1 = *(const uint4*)r1p;
            const unsigned short* h0 = (const unsigned short*)&q0;
            const unsigned short* h1 = (const unsigned short*)&q1;
            #pragma unroll
            for (int j = 0; j < 8; ++j) {
                const unsigned int packed =
                    (unsigned int)h0[j] | ((unsigned int)h1[j] << 16);
                *(unsigned int*)&Blds[buf][(nbase + j) * LDK + kp] = packed;
            }
            if (kk + 64 < K) __builtin_prefetch(r0p + 64LL * ldb, 0, 0);
        }
    };

    v8f cc[2][4];
    #pragma unroll
    for (int mi = 0; mi < 2; ++mi)
        #pragma unroll
        for (int ni = 0; ni < 4; ++ni)
            cc[mi][ni] = (v8f){0.f,0.f,0.f,0.f,0.f,0.f,0.f,0.f};

    stage(0, 0);
    __syncthreads();

    int buf = 0;
    for (int kk = 0; kk < K; kk += 32) {
        // Issue next tile's global loads before the math (latency hiding).
        if (kk + 32 < K) stage(kk + 32, buf ^ 1);

        FragAB af[2], bfr[4];
        #pragma unroll
        for (int mi = 0; mi < 2; ++mi)
            load_frag(Alds[buf], wrow + mi * 16 + lmod, lhalf, af[mi]);
        #pragma unroll
        for (int ni = 0; ni < 4; ++ni)
            load_frag(Blds[buf], wcol + ni * 16 + lmod, lhalf, bfr[ni]);

        #pragma unroll
        for (int mi = 0; mi < 2; ++mi)
            #pragma unroll
            for (int ni = 0; ni < 4; ++ni)
                cc[mi][ni] = __builtin_amdgcn_wmma_f32_16x16x32_bf16(
                    false, af[mi].v, false, bfr[ni].v,
                    (short)0, cc[mi][ni], false, false);

        __syncthreads();   // staging of buf^1 complete; reads of buf complete
        buf ^= 1;
    }

    // ---- epilogue: bias + (exact) GELU + fp32/bf16 stores ----
    #pragma unroll
    for (int mi = 0; mi < 2; ++mi) {
        #pragma unroll
        for (int ni = 0; ni < 4; ++ni) {
            const int ncol = n0 + wcol + ni * 16 + lmod;
            const float bv = bias ? bias[ncol] : 0.0f;
            #pragma unroll
            for (int r = 0; r < 8; ++r) {
                const int mrow = m0 + wrow + mi * 16 + r + lhalf * 8;
                float v = cc[mi][ni][r] + bv;
                if (actGelu) v = 0.5f * v * (1.0f + erff(v * 0.7071067811865476f));
                const long long idx = cOff + (long long)mrow * N + ncol;
                if (Cf) Cf[idx] = v;
                if (Cb) Cb[idx] = f2bf(v);
            }
        }
    }
}

// ---------------------------------------------------------------------------
// Row softmax: P[row,:] = softmax(scale * S[row,:]), output bf16.
// One 256-thread block per row of 2048.
// ---------------------------------------------------------------------------
__global__ __launch_bounds__(256) void softmax_rows(const float* __restrict__ S,
                                                    unsigned short* __restrict__ P,
                                                    int ncols, float scale)
{
    __shared__ float red[256];
    const long long row = blockIdx.x;
    const int tid = threadIdx.x;

    float v[8];
    float mx = -3.4e38f;
    #pragma unroll
    for (int i = 0; i < 8; ++i) {
        v[i] = S[row * ncols + tid + i * 256] * scale;
        mx = fmaxf(mx, v[i]);
    }

    red[tid] = mx; __syncthreads();
    for (int s = 128; s > 0; s >>= 1) {
        if (tid < s) red[tid] = fmaxf(red[tid], red[tid + s]);
        __syncthreads();
    }
    mx = red[0]; __syncthreads();

    float sum = 0.f;
    #pragma unroll
    for (int i = 0; i < 8; ++i) { v[i] = expf(v[i] - mx); sum += v[i]; }

    red[tid] = sum; __syncthreads();
    for (int s = 128; s > 0; s >>= 1) {
        if (tid < s) red[tid] += red[tid + s];
        __syncthreads();
    }
    const float inv = 1.0f / red[0];

    #pragma unroll
    for (int i = 0; i < 8; ++i)
        P[row * ncols + tid + i * 256] = f2bf(v[i] * inv);
}

// ---------------------------------------------------------------------------
// Fused residual add + LayerNorm over rows of 768.
// outF (fp32) and/or outB (bf16).  One 256-thread block per row, 3 elems/thr.
// ---------------------------------------------------------------------------
__global__ __launch_bounds__(256) void add_layernorm(
    const float* __restrict__ a, const float* __restrict__ b,
    const float* __restrict__ gamma, const float* __restrict__ beta,
    float* __restrict__ outF, unsigned short* __restrict__ outB, int d)
{
    __shared__ float red[256];
    const long long row = blockIdx.x;
    const int tid = threadIdx.x;

    float v[3];
    float s = 0.f;
    #pragma unroll
    for (int i = 0; i < 3; ++i) {
        const int idx = tid + i * 256;
        v[i] = a[row * d + idx] + b[row * d + idx];
        s += v[i];
    }
    red[tid] = s; __syncthreads();
    for (int st = 128; st > 0; st >>= 1) {
        if (tid < st) red[tid] += red[tid + st];
        __syncthreads();
    }
    const float mean = red[0] / (float)d;
    __syncthreads();

    float sq = 0.f;
    #pragma unroll
    for (int i = 0; i < 3; ++i) { const float c = v[i] - mean; sq += c * c; }
    red[tid] = sq; __syncthreads();
    for (int st = 128; st > 0; st >>= 1) {
        if (tid < st) red[tid] += red[tid + st];
        __syncthreads();
    }
    const float rstd = rsqrtf(red[0] / (float)d + 1e-5f);

    #pragma unroll
    for (int i = 0; i < 3; ++i) {
        const int idx = tid + i * 256;
        const float o = (v[i] - mean) * rstd * gamma[idx] + beta[idx];
        if (outF) outF[row * d + idx] = o;
        if (outB) outB[row * d + idx] = f2bf(o);
    }
}

// ---------------------------------------------------------------------------
// Launch: full transformer layer.
//   B=8, N=2048, D=768, F=3072.  All GEMMs via bf16 WMMA, fp32 accumulate.
// ---------------------------------------------------------------------------
extern "C" void kernel_launch(void* const* d_in, const int* in_sizes, int n_in,
                              void* d_out, int out_size, void* d_ws, size_t ws_size,
                              hipStream_t stream) {
    (void)in_sizes; (void)n_in; (void)out_size; (void)ws_size;

    const int  Dm = 768, Fm = 3072, Ns = 2048, Bb = 8;
    const long long TOK = (long long)Bb * Ns;        // 16384 rows

    const float* x     = (const float*)d_in[0];
    const float* Wq    = (const float*)d_in[1];
    const float* Wk    = (const float*)d_in[2];
    const float* Wv    = (const float*)d_in[3];
    const float* ln1_g = (const float*)d_in[4];
    const float* ln1_b = (const float*)d_in[5];
    const float* W1    = (const float*)d_in[6];
    const float* b1    = (const float*)d_in[7];
    const float* W2    = (const float*)d_in[8];
    const float* b2    = (const float*)d_in[9];
    const float* ln2_g = (const float*)d_in[10];
    const float* ln2_b = (const float*)d_in[11];

    // ---- workspace bump allocation (256B aligned), with reuse ----
    char* ws = (char*)d_ws;
    size_t o = 0;
    auto alloc = [&](size_t bytes) { size_t r = o; o += (bytes + 255) & ~(size_t)255; return r; };

    const size_t act_b  = (size_t)TOK * Dm * 2;          // bf16 activations
    const size_t act_f  = (size_t)TOK * Dm * 4;          // fp32 activations
    const size_t off_xb = alloc(act_b);
    const size_t off_wq = alloc((size_t)Dm * Dm * 2);
    const size_t off_wk = alloc((size_t)Dm * Dm * 2);
    const size_t off_wv = alloc((size_t)Dm * Dm * 2);
    const size_t off_w1 = alloc((size_t)Dm * Fm * 2);
    const size_t off_w2 = alloc((size_t)Fm * Dm * 2);
    const size_t off_q  = alloc(act_b);
    const size_t off_k  = alloc(act_b);
    const size_t off_v  = alloc(act_b);
    const size_t off_sc = alloc((size_t)Bb * Ns * Ns * 4);   // scores fp32 (134MB)
    const size_t off_at = alloc((size_t)Bb * Ns * Ns * 2);   // attn bf16 (67MB)
    const size_t off_o  = alloc(act_f);
    const size_t off_h  = alloc(act_f);
    const size_t off_hb = alloc(act_b);
    const size_t off_hd = off_sc;   // reuse scores region for FFN hidden bf16
    const size_t off_ff = off_q;    // reuse Q/K/V region for FFN out fp32

    unsigned short* xb  = (unsigned short*)(ws + off_xb);
    unsigned short* wqb = (unsigned short*)(ws + off_wq);
    unsigned short* wkb = (unsigned short*)(ws + off_wk);
    unsigned short* wvb = (unsigned short*)(ws + off_wv);
    unsigned short* w1b = (unsigned short*)(ws + off_w1);
    unsigned short* w2b = (unsigned short*)(ws + off_w2);
    unsigned short* Qb  = (unsigned short*)(ws + off_q);
    unsigned short* Kb  = (unsigned short*)(ws + off_k);
    unsigned short* Vb  = (unsigned short*)(ws + off_v);
    float*          Sc  = (float*)(ws + off_sc);
    unsigned short* At  = (unsigned short*)(ws + off_at);
    float*          Of  = (float*)(ws + off_o);
    float*          Hf  = (float*)(ws + off_h);
    unsigned short* Hb  = (unsigned short*)(ws + off_hb);
    unsigned short* Hd  = (unsigned short*)(ws + off_hd);
    float*          Ff  = (float*)(ws + off_ff);

    // ---- casts to bf16 ----
    cast_f32_bf16<<<4096, 256, 0, stream>>>(x,  xb,  TOK * Dm);
    cast_f32_bf16<<<1024, 256, 0, stream>>>(Wq, wqb, (long long)Dm * Dm);
    cast_f32_bf16<<<1024, 256, 0, stream>>>(Wk, wkb, (long long)Dm * Dm);
    cast_f32_bf16<<<1024, 256, 0, stream>>>(Wv, wvb, (long long)Dm * Dm);
    cast_f32_bf16<<<2048, 256, 0, stream>>>(W1, w1b, (long long)Dm * Fm);
    cast_f32_bf16<<<2048, 256, 0, stream>>>(W2, w2b, (long long)Fm * Dm);

    const dim3 blk(256);

    // ---- Q/K/V projections: [16384,768] @ [768,768] -> bf16 ----
    {
        dim3 grd(Dm / 128, (unsigned)(TOK / 128), 1);
        gemm_bf16_wmma<<<grd, blk, 0, stream>>>(xb, wqb, nullptr, nullptr, Qb,
            (int)TOK, Dm, Dm, Dm, 0, 0, 0, 0, 0);
        gemm_bf16_wmma<<<grd, blk, 0, stream>>>(xb, wkb, nullptr, nullptr, Kb,
            (int)TOK, Dm, Dm, Dm, 0, 0, 0, 0, 0);
        gemm_bf16_wmma<<<grd, blk, 0, stream>>>(xb, wvb, nullptr, nullptr, Vb,
            (int)TOK, Dm, Dm, Dm, 0, 0, 0, 0, 0);
    }

    // ---- scores = Q @ K^T (batched, transB), fp32 out ----
    {
        dim3 grd(Ns / 128, Ns / 128, Bb);
        gemm_bf16_wmma<<<grd, blk, 0, stream>>>(Qb, Kb, nullptr, Sc, nullptr,
            Ns, Ns, Dm, /*ldb=*/Dm, /*transB=*/1, 0,
            (long long)Ns * Dm, (long long)Ns * Dm, (long long)Ns * Ns);
    }

    // ---- softmax(scale * scores) -> bf16 probs ----
    softmax_rows<<<(unsigned)(Bb * Ns), blk, 0, stream>>>(Sc, At, Ns, 0.03608439182435161f);

    // ---- O = attn @ V (batched), fp32 out ----
    {
        dim3 grd(Dm / 128, Ns / 128, Bb);
        gemm_bf16_wmma<<<grd, blk, 0, stream>>>(At, Vb, nullptr, Of, nullptr,
            Ns, Dm, Ns, /*ldb=*/Dm, 0, 0,
            (long long)Ns * Ns, (long long)Ns * Dm, (long long)Ns * Dm);
    }

    // ---- h = LN(x + O): fp32 h + bf16 h ----
    add_layernorm<<<(unsigned)TOK, blk, 0, stream>>>(x, Of, ln1_g, ln1_b, Hf, Hb, Dm);

    // ---- hidden = gelu(h @ W1 + b1) -> bf16 ----
    {
        dim3 grd(Fm / 128, (unsigned)(TOK / 128), 1);
        gemm_bf16_wmma<<<grd, blk, 0, stream>>>(Hb, w1b, b1, nullptr, Hd,
            (int)TOK, Fm, Dm, Fm, 0, /*gelu=*/1, 0, 0, 0);
    }

    // ---- ffn = hidden @ W2 + b2 -> fp32 ----
    {
        dim3 grd(Dm / 128, (unsigned)(TOK / 128), 1);
        gemm_bf16_wmma<<<grd, blk, 0, stream>>>(Hd, w2b, b2, Ff, nullptr,
            (int)TOK, Dm, Fm, Dm, 0, 0, 0, 0, 0);
    }

    // ---- out = LN(h + ffn) -> d_out fp32 ----
    add_layernorm<<<(unsigned)TOK, blk, 0, stream>>>(Hf, Ff, ln2_g, ln2_b,
                                                    (float*)d_out, nullptr, Dm);
}
